// CrossModalAttention_14886356648201
// MI455X (gfx1250) — matrix operand order repaired and checked by reading, not compiled, not run on hardware
//
#include <hip/hip_runtime.h>

typedef __attribute__((ext_vector_type(16))) _Float16 v16h;
typedef __attribute__((ext_vector_type(8)))  float    v8f;

namespace {

constexpr int kC = 256;    // channels
constexpr int kN = 4096;   // H*W tokens
constexpr int kB = 2;      // batch
constexpr int kD = 32;     // head dim
constexpr float kScale = 0.17677669529663687f;  // 32^-0.5

union V16HU { uint4 u[2]; v16h h; };

__device__ inline v16h ld_v16h(const _Float16* p0, const _Float16* p1) {
  V16HU r;
  r.u[0] = *reinterpret_cast<const uint4*>(p0);
  r.u[1] = *reinterpret_cast<const uint4*>(p1);
  return r.h;
}

// A-operand from f32 memory: halves 0-7 <- p[0..7], halves 8-15 <- p[8..15]
__device__ inline v16h ld_a_f32(const float* p) {
  float4 f0 = *reinterpret_cast<const float4*>(p);
  float4 f1 = *reinterpret_cast<const float4*>(p + 4);
  float4 f2 = *reinterpret_cast<const float4*>(p + 8);
  float4 f3 = *reinterpret_cast<const float4*>(p + 12);
  v16h r;
  r[0] = (_Float16)f0.x;  r[1] = (_Float16)f0.y;
  r[2] = (_Float16)f0.z;  r[3] = (_Float16)f0.w;
  r[4] = (_Float16)f1.x;  r[5] = (_Float16)f1.y;
  r[6] = (_Float16)f1.z;  r[7] = (_Float16)f1.w;
  r[8] = (_Float16)f2.x;  r[9] = (_Float16)f2.y;
  r[10] = (_Float16)f2.z; r[11] = (_Float16)f2.w;
  r[12] = (_Float16)f3.x; r[13] = (_Float16)f3.y;
  r[14] = (_Float16)f3.z; r[15] = (_Float16)f3.w;
  return r;
}

__device__ inline v8f v8f_zero() {
  v8f z = {0.f, 0.f, 0.f, 0.f, 0.f, 0.f, 0.f, 0.f};
  return z;
}

__device__ inline v8f wmma_f16(v16h a, v16h b, v8f c) {
  return __builtin_amdgcn_wmma_f32_16x16x32_f16(
      false, a, false, b, (short)0, c, false, false);
}

// ---------------------------------------------------------------------------
// Kernel 1: six projections, f32 -> f16.
//   p: 0=Q1(wq,spatial) 1=K1(wk,freq) 2=V1(wv,freq)
//      3=Q2(wq,freq)    4=K2(wk,spatial) 5=V2(wv,spatial)
// Q/K slots (0,1,3,4) written TRANSPOSED (n, c) with vector b128 stores.
// V slots (2,5) written normal (c, n).
// ---------------------------------------------------------------------------
__global__ __launch_bounds__(128) void proj_kernel(
    const float* __restrict__ spatial, const float* __restrict__ freq,
    const float* __restrict__ wq, const float* __restrict__ bq,
    const float* __restrict__ wk, const float* __restrict__ bk,
    const float* __restrict__ wv, const float* __restrict__ bv,
    _Float16* __restrict__ proj) {
  const int nbase = blockIdx.x * 64;
  const int mbase = blockIdx.y * 64;
  const int pz = blockIdx.z;  // p * kB + b
  const int p = pz / kB, b = pz % kB;

  const float* W; const float* bias; const float* X;
  switch (p) {
    case 0:  W = wq; bias = bq; X = spatial; break;
    case 1:  W = wk; bias = bk; X = freq;    break;
    case 2:  W = wv; bias = bv; X = freq;    break;
    case 3:  W = wq; bias = bq; X = freq;    break;
    case 4:  W = wk; bias = bk; X = spatial; break;
    default: W = wv; bias = bv; X = spatial; break;
  }
  X += (size_t)b * kC * kN;
  _Float16* Y = proj + (size_t)pz * kC * kN;

  __shared__ alignas(16) _Float16 ldsX[64 * 32];  // [n][k] transposed B-side

  const int tid  = threadIdx.x;
  const int lane = tid & 31;
  const int wave = tid >> 5;
  const int l16  = lane & 15;
  const bool hi  = lane >= 16;
  const int ak   = hi ? 16 : 0;
  const int bk0  = hi ? 16 : 0;
  const int arow = mbase + wave * 16 + l16;  // A row this lane owns

  v8f acc[4];
  for (int t = 0; t < 4; ++t) acc[t] = v8f_zero();

  for (int k0 = 0; k0 < kC; k0 += 32) {
    __syncthreads();
    // Stage X tile transposed: float4 loads along n, f16 scatter along c.
#pragma unroll
    for (int it = 0; it < 4; ++it) {
      int idx4 = tid + it * 128;              // 512 float4 = 32x64 f32
      int c = idx4 >> 4, nq = (idx4 & 15) << 2;
      float4 f = *reinterpret_cast<const float4*>(
          &X[(size_t)(k0 + c) * kN + nbase + nq]);
      ldsX[(nq + 0) * 32 + c] = (_Float16)f.x;
      ldsX[(nq + 1) * 32 + c] = (_Float16)f.y;
      ldsX[(nq + 2) * 32 + c] = (_Float16)f.z;
      ldsX[(nq + 3) * 32 + c] = (_Float16)f.w;
    }
    __syncthreads();

    // A tile: straight from global W (f32 -> f16 in regs).
    v16h a = ld_a_f32(&W[(size_t)arow * kC + k0 + ak]);

#pragma unroll
    for (int t = 0; t < 4; ++t) {
      const int bn = t * 16 + l16;
      v16h bb = ld_v16h(&ldsX[bn * 32 + bk0], &ldsX[bn * 32 + bk0 + 8]);
      acc[t] = wmma_f16(a, bb, acc[t]);
    }
  }

  if (p != 2 && p != 5) {
    // Transposed (n, c) epilogue: 8 rows per lane are contiguous -> b128.
    const int mcb = mbase + wave * 16 + (hi ? 8 : 0);
    float bv8[8];
#pragma unroll
    for (int r = 0; r < 8; ++r) bv8[r] = bias[mcb + r];
#pragma unroll
    for (int t = 0; t < 4; ++t) {
      const int n = nbase + t * 16 + l16;
      union { uint4 u; _Float16 h[8]; } pk;
#pragma unroll
      for (int r = 0; r < 8; ++r) pk.h[r] = (_Float16)(acc[t][r] + bv8[r]);
      *reinterpret_cast<uint4*>(&Y[(size_t)n * kC + mcb]) = pk.u;
    }
  } else {
    // Normal (c, n) epilogue for V.
#pragma unroll
    for (int t = 0; t < 4; ++t) {
      const int n = nbase + t * 16 + l16;
#pragma unroll
      for (int r = 0; r < 8; ++r) {
        const int m = mbase + wave * 16 + (hi ? r + 8 : r);
        Y[(size_t)m * kN + n] = (_Float16)(acc[t][r] + bias[m]);
      }
    }
  }
}

// ---------------------------------------------------------------------------
// Kernel 2: flash attention, barrier-free. 4 independent waves per block,
// each owns 16 query rows. Q/K read from transposed (n,c) slots, V from
// (c,n). Output written transposed (n,c). LDS only for per-wave P relayout.
// ---------------------------------------------------------------------------
__global__ __launch_bounds__(128) void attn_kernel(
    const _Float16* __restrict__ proj, _Float16* __restrict__ attn) {
  const int mbase = blockIdx.x * 64;
  const int h = blockIdx.y;
  const int a = blockIdx.z / kB, b = blockIdx.z % kB;

  const _Float16* Qt = proj + (size_t)((a ? 3 : 0) * kB + b) * kC * kN;  // (n,c)
  const _Float16* Kt = proj + (size_t)((a ? 4 : 1) * kB + b) * kC * kN;  // (n,c)
  const _Float16* Vh = proj + (size_t)((a ? 5 : 2) * kB + b) * kC * kN +
                       (size_t)(h * kD) * kN;                            // (d,n)
  _Float16* Ot = attn + (size_t)(a * kB + b) * kN * kC + (size_t)h * kD; // (n,c)

  __shared__ alignas(16) _Float16 ldsP[4][16 * 64];  // per-wave [m][n]

  const int tid  = threadIdx.x;
  const int lane = tid & 31;
  const int wave = tid >> 5;
  const int l16  = lane & 15;
  const bool hi  = lane >= 16;
  const int ak   = hi ? 16 : 0;
  const int bk0  = hi ? 16 : 0;

  // Loop-invariant A operand: Q^T rows straight from global.
  const int qrow = mbase + wave * 16 + l16;
  const _Float16* qp = Qt + (size_t)qrow * kC + h * kD + ak;
  const v16h qa = ld_v16h(qp, qp + 8);

  float Mrow[8], Srow[8];
#pragma unroll
  for (int r = 0; r < 8; ++r) { Mrow[r] = -1e30f; Srow[r] = 0.f; }
  v8f o0 = v8f_zero(), o1 = v8f_zero();
  _Float16* pb = &ldsP[wave][0];

  for (int n0 = 0; n0 < kN; n0 += 64) {
    if (n0 + 64 < kN)
      __builtin_prefetch(Kt + (size_t)(n0 + 64 + lane) * kC + h * kD, 0, 1);

    // Scores: 4 WMMAs, K B-tiles straight from global (n,c) layout.
    v8f s[4];
#pragma unroll
    for (int t = 0; t < 4; ++t) {
      const _Float16* kp = Kt + (size_t)(n0 + t * 16 + l16) * kC + h * kD + bk0;
      v16h kb = ld_v16h(kp, kp + 8);
      s[t] = wmma_f16(qa, kb, v8f_zero());
    }

    // Online softmax over 64 keys.
#pragma unroll
    for (int r = 0; r < 8; ++r) {
      float x[4];
      float mx = -1e30f;
#pragma unroll
      for (int t = 0; t < 4; ++t) {
        x[t] = s[t][r] * kScale;
        mx = fmaxf(mx, x[t]);
      }
      mx = fmaxf(mx, __shfl_xor(mx, 1, 32));
      mx = fmaxf(mx, __shfl_xor(mx, 2, 32));
      mx = fmaxf(mx, __shfl_xor(mx, 4, 32));
      mx = fmaxf(mx, __shfl_xor(mx, 8, 32));
      float newM = fmaxf(Mrow[r], mx);
      float corr = __expf(Mrow[r] - newM);
      float ts = 0.f;
#pragma unroll
      for (int t = 0; t < 4; ++t) {
        x[t] = __expf(x[t] - newM);
        ts += x[t];
      }
      ts += __shfl_xor(ts, 1, 32);
      ts += __shfl_xor(ts, 2, 32);
      ts += __shfl_xor(ts, 4, 32);
      ts += __shfl_xor(ts, 8, 32);
      Srow[r] = Srow[r] * corr + ts;
      Mrow[r] = newM;
      o0[r] *= corr;
      o1[r] *= corr;
      const int m = hi ? r + 8 : r;
#pragma unroll
      for (int t = 0; t < 4; ++t)
        pb[m * 64 + t * 16 + l16] = (_Float16)x[t];
    }

    // O^T += P * V^T : 2 K-chunks of 32 keys, V B-tiles straight from global.
#pragma unroll
    for (int kk = 0; kk < 2; ++kk) {
      v16h pa = ld_v16h(&pb[l16 * 64 + kk * 32 + ak],
                        &pb[l16 * 64 + kk * 32 + ak + 8]);
      const _Float16* vp0 = Vh + (size_t)l16 * kN + n0 + kk * 32 + bk0;
      const _Float16* vp1 = Vh + (size_t)(16 + l16) * kN + n0 + kk * 32 + bk0;
      o0 = wmma_f16(pa, ld_v16h(vp0, vp0 + 8), o0);
      o1 = wmma_f16(pa, ld_v16h(vp1, vp1 + 8), o1);
    }
  }

  // Normalize and store transposed (n, c).
#pragma unroll
  for (int r = 0; r < 8; ++r) {
    float inv = 1.0f / Srow[r];
    const int mrow = mbase + wave * 16 + (hi ? r + 8 : r);
    Ot[(size_t)mrow * kC + l16]      = (_Float16)(o0[r] * inv);
    Ot[(size_t)mrow * kC + 16 + l16] = (_Float16)(o1[r] * inv);
  }
}

// ---------------------------------------------------------------------------
// Kernel 3: out = spatial + freq + wo @ (A1 + A2) + 2*bo. LDS-free:
// A-operand = wo rows (f32->f16 in regs), B-operand = A1t+A2t from
// transposed (n,c) buffers via b128 loads + packed f16 adds.
// ---------------------------------------------------------------------------
__global__ __launch_bounds__(128) void out_proj_kernel(
    const float* __restrict__ spatial, const float* __restrict__ freq,
    const float* __restrict__ wo, const float* __restrict__ bo,
    const _Float16* __restrict__ attn, float* __restrict__ out) {
  const int nbase = blockIdx.x * 64;
  const int mbase = blockIdx.y * 64;
  const int b = blockIdx.z;

  const _Float16* A1 = attn + (size_t)b * kN * kC;         // (n,c)
  const _Float16* A2 = attn + (size_t)(kB + b) * kN * kC;  // (n,c)
  const float* sp = spatial + (size_t)b * kC * kN;
  const float* fr = freq + (size_t)b * kC * kN;

  const int tid  = threadIdx.x;
  const int lane = tid & 31;
  const int wave = tid >> 5;
  const int l16  = lane & 15;
  const bool hi  = lane >= 16;
  const int ak   = hi ? 16 : 0;
  const int bk0  = hi ? 16 : 0;
  const int arow = mbase + wave * 16 + l16;

  v8f acc[4];
  for (int t = 0; t < 4; ++t) acc[t] = v8f_zero();

  for (int k0 = 0; k0 < kC; k0 += 32) {
    v16h a = ld_a_f32(&wo[(size_t)arow * kC + k0 + ak]);
#pragma unroll
    for (int t = 0; t < 4; ++t) {
      const size_t boff = (size_t)(nbase + t * 16 + l16) * kC + k0 + bk0;
      v16h x1 = ld_v16h(&A1[boff], &A1[boff] + 8);
      v16h x2 = ld_v16h(&A2[boff], &A2[boff] + 8);
      v16h xs = x1 + x2;  // v_pk_add_f16
      acc[t] = wmma_f16(a, xs, acc[t]);
    }
  }

#pragma unroll
  for (int t = 0; t < 4; ++t) {
    const int n = nbase + t * 16 + l16;
#pragma unroll
    for (int r = 0; r < 8; ++r) {
      const int m = mbase + wave * 16 + (hi ? r + 8 : r);
      const size_t idx = (size_t)m * kN + n;
      out[(size_t)b * kC * kN + idx] =
          acc[t][r] + 2.0f * bo[m] + sp[idx] + fr[idx];
    }
  }
}

}  // namespace

extern "C" void kernel_launch(void* const* d_in, const int* in_sizes, int n_in,
                              void* d_out, int out_size, void* d_ws, size_t ws_size,
                              hipStream_t stream) {
  (void)in_sizes; (void)n_in; (void)out_size; (void)ws_size;
  const float* spatial = (const float*)d_in[0];
  const float* freq    = (const float*)d_in[1];
  const float* wq = (const float*)d_in[2];
  const float* bq = (const float*)d_in[3];
  const float* wk = (const float*)d_in[4];
  const float* bk = (const float*)d_in[5];
  const float* wv = (const float*)d_in[6];
  const float* bv = (const float*)d_in[7];
  const float* wo = (const float*)d_in[8];
  const float* bo = (const float*)d_in[9];
  float* out = (float*)d_out;

  _Float16* wsh  = (_Float16*)d_ws;
  _Float16* proj = wsh;                             // 12 * C * N halves
  _Float16* attn = wsh + (size_t)6 * kB * kC * kN;  //  4 * C * N halves

  dim3 g1(kN / 64, kC / 64, 6 * kB);
  proj_kernel<<<g1, 128, 0, stream>>>(spatial, freq, wq, bq, wk, bk, wv, bv, proj);

  dim3 g2(kN / 64, 8, 2 * kB);
  attn_kernel<<<g2, 128, 0, stream>>>(proj, attn);

  dim3 g3(kN / 64, kC / 64, kB);
  out_proj_kernel<<<g3, 128, 0, stream>>>(spatial, freq, wo, bo, attn, out);
}
